// LocalGlobalAttention_39470749450368
// MI455X (gfx1250) — matrix operand order repaired and checked by reading, not compiled, not run on hardware
//
#include <hip/hip_runtime.h>

typedef unsigned short u16;
typedef __attribute__((ext_vector_type(16))) __bf16 v16bf;
typedef __attribute__((ext_vector_type(8)))  float  v8f;
typedef __attribute__((ext_vector_type(8)))  u16    v8u;
typedef __attribute__((ext_vector_type(16))) u16    v16u;
typedef int g4i __attribute__((__vector_size__(16)));   // matches builtin pointee

#define BATCH 2
#define NSEQ  4096
#define TOK   8192      // BATCH*NSEQ
#define CDIM  1024
#define C3    3072
#define HEADS 16
#define DH    64
#define NB    64        // blocks per sequence (N/W)
#define NEGF  (-1.0e9f)

// ---- optional CDNA5 async global->LDS path (guarded; falls back cleanly) ----
#if defined(__has_builtin)
#  if __has_builtin(__builtin_amdgcn_global_load_async_to_lds_b128) && \
      __has_builtin(__builtin_amdgcn_s_wait_asynccnt)
#    define USE_ASYNC_LDS 1
#  endif
#endif
#ifndef USE_ASYNC_LDS
#  define USE_ASYNC_LDS 0
#endif

__device__ __forceinline__ void stage_chunk16(const u16* gp, u16* lp) {
#if USE_ASYNC_LDS
  __builtin_amdgcn_global_load_async_to_lds_b128(
      (__attribute__((address_space(1))) g4i*)gp,
      (__attribute__((address_space(3))) g4i*)lp,
      0, 0);
#else
  *(v8u*)lp = *(const v8u*)gp;
#endif
}
__device__ __forceinline__ void wait_stage() {
#if USE_ASYNC_LDS
  __builtin_amdgcn_s_wait_asynccnt(0);
#endif
}

// ---------------- helpers ----------------

__device__ __forceinline__ u16 f2bf(float f) {
  unsigned u = __builtin_bit_cast(unsigned, f);
  u += 0x7FFFu + ((u >> 16) & 1u);           // round-to-nearest-even
  return (u16)(u >> 16);
}

__device__ __forceinline__ v8u zero8() {
  v8u z;
#pragma unroll
  for (int i = 0; i < 8; ++i) z[i] = 0;
  return z;
}

__device__ __forceinline__ v16bf pack16(v8u lo, v8u hi) {
  v16u t;
#pragma unroll
  for (int i = 0; i < 8; ++i) { t[i] = lo[i]; t[i + 8] = hi[i]; }
  return __builtin_bit_cast(v16bf, t);
}

// A fragment (16x32 bf16, row-major M x K): elems 0..15 -> K = half*16+e (contiguous)
__device__ __forceinline__ v16bf load_fragA(const u16* p, int half) {
  v8u lo = *(const v8u*)(p + half * 16);
  v8u hi = *(const v8u*)(p + half * 16 + 8);
  return pack16(lo, hi);
}
// B fragment (32x16 bf16, stored N x K): elems 0..15 -> K = half*16+e (contiguous)
__device__ __forceinline__ v16bf load_fragB(const u16* p, int half) {
  v8u lo = *(const v8u*)(p + half * 16);
  v8u hi = *(const v8u*)(p + half * 16 + 8);
  return pack16(lo, hi);
}
__device__ __forceinline__ v16bf load_fragB_guard(const u16* p, int half, bool valid) {
  v8u lo = valid ? *(const v8u*)(p + half * 16) : zero8();
  v8u hi = valid ? *(const v8u*)(p + half * 16 + 8) : zero8();
  return pack16(lo, hi);
}
__device__ __forceinline__ v16bf zerofrag() { return pack16(zero8(), zero8()); }

__device__ __forceinline__ v8f wmma_bf16(v16bf a, v16bf b, v8f c) {
  return __builtin_amdgcn_wmma_f32_16x16x32_bf16(false, a, false, b, (short)0, c, false, false);
}

__device__ __forceinline__ v8f zeroacc() {
  v8f z;
#pragma unroll
  for (int i = 0; i < 8; ++i) z[i] = 0.0f;
  return z;
}

// ---------------- stage 0: conversions ----------------

__global__ void k_f2bf(const float* __restrict__ in, u16* __restrict__ out, long n) {
  long i = (long)blockIdx.x * blockDim.x + threadIdx.x;
  long stride = (long)gridDim.x * blockDim.x;
  for (; i < n; i += stride) out[i] = f2bf(in[i]);
}

// out[NO][KO] = transpose(in[KO][NO]), f32 -> bf16
__global__ void k_tr_f2bf(const float* __restrict__ in, u16* __restrict__ out, int NO, int KO) {
  long n = (long)NO * KO;
  long i = (long)blockIdx.x * blockDim.x + threadIdx.x;
  long stride = (long)gridDim.x * blockDim.x;
  for (; i < n; i += stride) {
    int r = (int)(i / KO);
    int c = (int)(i % KO);
    out[i] = f2bf(in[(long)c * NO + r]);
  }
}

// ---------------- stage 1: QKV GEMM (8192x1024)x(1024x3072) ----------------
// 4-wave block computes a 64x256 strip: shared A k-slice double-buffered in LDS
// (async global->LDS when available), each wave owns a 64-column slice.

__global__ __launch_bounds__(128) void k_gemm_qkv(
    const u16* __restrict__ Xb, const u16* __restrict__ Wqt,
    u16* __restrict__ Qb, u16* __restrict__ Kb,
    u16* __restrict__ Vt, u16* __restrict__ Vgt) {
  __shared__ u16 Atile[2][64 * 32];        // 2 x 4KB double buffer

  const int bx = blockIdx.x;               // 128 * 12
  const int mg = bx / 12, ng = bx % 12;
  const int tid = threadIdx.x;
  const int wave = tid >> 5, lane = tid & 31;
  const int half = lane >> 4, nidx = lane & 15;
  const int m0 = mg * 64;
  const int n0 = ng * 256 + wave * 64;

  v8f acc[4][4];
#pragma unroll
  for (int i = 0; i < 4; ++i)
#pragma unroll
    for (int j = 0; j < 4; ++j) acc[i][j] = zeroacc();

  // stage one 64x32 A slice (4KB = 256 x 16B chunks, 2 per thread)
  auto stage = [&](int buf, int kb) {
#pragma unroll
    for (int q = 0; q < 2; ++q) {
      const int chunk = tid + q * 128;
      const int row = chunk >> 2, cc = chunk & 3;
      stage_chunk16(Xb + (size_t)(m0 + row) * CDIM + kb + cc * 8,
                    &Atile[buf][chunk * 8]);
    }
  };

  stage(0, 0);
  wait_stage();
  __syncthreads();

  for (int kb = 0; kb < CDIM; kb += 32) {
    const int buf = (kb >> 5) & 1;
    if (kb + 32 < CDIM) stage(buf ^ 1, kb + 32);

    v16bf a[4], b[4];
#pragma unroll
    for (int i = 0; i < 4; ++i)
      a[i] = load_fragA(&Atile[buf][(i * 16 + nidx) * 32], half);
#pragma unroll
    for (int j = 0; j < 4; ++j)
      b[j] = load_fragB(Wqt + (size_t)(n0 + j * 16 + nidx) * CDIM + kb, half);
#pragma unroll
    for (int i = 0; i < 4; ++i)
#pragma unroll
      for (int j = 0; j < 4; ++j) acc[i][j] = wmma_bf16(a[i], b[j], acc[i][j]);

    wait_stage();
    __syncthreads();
  }

  const float scale = 0.125f;              // Dh^-0.5 folded into Q
#pragma unroll
  for (int i = 0; i < 4; ++i)
#pragma unroll
    for (int j = 0; j < 4; ++j)
#pragma unroll
      for (int r = 0; r < 8; ++r) {
        const int m = m0 + i * 16 + r + 8 * half;   // token row in [0,8192)
        const int c = n0 + j * 16 + nidx;           // col in [0,3072)
        const float v = acc[i][j][r];
        const int s = c >> 10;                      // 0=q 1=k 2=v
        const int h = (c >> 6) & 15;
        const int d = c & 63;
        const int bb = m >> 12;
        const int nn = m & (NSEQ - 1);
        const size_t bh = (size_t)(bb * HEADS + h);
        if (s == 0) {
          Qb[(bh * NSEQ + nn) * DH + d] = f2bf(v * scale);
        } else if (s == 1) {
          Kb[(bh * NSEQ + nn) * DH + d] = f2bf(v);
        } else {
          const u16 bv = f2bf(v);
          Vt[(bh * DH + d) * NSEQ + nn] = bv;
          if ((nn & 63) == 0) Vgt[(bh * DH + d) * NB + (nn >> 6)] = bv;
        }
      }
}

// ---------------- stage 2: local + global attention ----------------
// one block (4 waves) per (batch, head, query-block of 64)

__global__ __launch_bounds__(128) void k_attn(
    const u16* __restrict__ Qb, const u16* __restrict__ Kb,
    const u16* __restrict__ Vt, const u16* __restrict__ Vgt,
    u16* __restrict__ AO) {
  __shared__ float Llog[64][192];   // 48 KB local logits -> probs (bf16 in place)
  __shared__ float Glog[64][64];    // 16 KB global logits -> probs

  const int id = blockIdx.x;               // BATCH*HEADS*NB
  const int blk = id & (NB - 1);
  const int h = (id >> 6) & (HEADS - 1);
  const int bb = id >> 10;
  const size_t bh = (size_t)(bb * HEADS + h);

  const int wave = threadIdx.x >> 5;
  const int lane = threadIdx.x & 31;
  const int half = lane >> 4, nidx = lane & 15;
  const int q0 = wave * 16;

  const u16* Qbase = Qb + (bh * NSEQ + (size_t)blk * 64) * DH;
  const u16* Kbase = Kb + bh * NSEQ * DH;

  // --- logits via WMMA (Q rows x {192 local keys, 64 global keys}) ---
  v8f accL[12], accG[4];
#pragma unroll
  for (int j = 0; j < 12; ++j) accL[j] = zeroacc();
#pragma unroll
  for (int j = 0; j < 4; ++j) accG[j] = zeroacc();

#pragma unroll
  for (int ks = 0; ks < 2; ++ks) {          // Dh = 64 = 2 * 32
    const v16bf a = load_fragA(Qbase + (size_t)(q0 + nidx) * DH + ks * 32, half);
#pragma unroll
    for (int j = 0; j < 12; ++j) {
      const int t = blk * 64 - 64 + j * 16 + nidx;   // neighborhood token
      v16bf bf;
      if (t >= 0 && t < NSEQ)
        bf = load_fragB(Kbase + (size_t)t * DH + ks * 32, half);
      else
        bf = zerofrag();
      accL[j] = wmma_bf16(a, bf, accL[j]);
    }
#pragma unroll
    for (int j = 0; j < 4; ++j) {
      const int g = j * 16 + nidx;                   // global key -> token g*64
      const v16bf bf = load_fragB(Kbase + (size_t)g * 64 * DH + ks * 32, half);
      accG[j] = wmma_bf16(a, bf, accG[j]);
    }
  }

#pragma unroll
  for (int j = 0; j < 12; ++j)
#pragma unroll
    for (int r = 0; r < 8; ++r)
      Llog[q0 + r + 8 * half][j * 16 + nidx] = accL[j][r];
#pragma unroll
  for (int j = 0; j < 4; ++j)
#pragma unroll
    for (int r = 0; r < 8; ++r)
      Glog[q0 + r + 8 * half][j * 16 + nidx] = accG[j][r];

  __syncthreads();

  // --- per-row masked softmax (f32), write bf16 probs in place ---
  const int tid = threadIdx.x;
  if (tid < 64) {
    const int q = tid;
    float mx = -3.4e38f;
    for (int kk = 0; kk < 192; ++kk) {
      const int rel = kk - 64 - q;
      const int jab = blk * 64 + kk - 64;
      const bool valid = (rel >= -32) && (rel <= 31) && (jab >= 0) && (jab < NSEQ);
      const float v = valid ? Llog[q][kk] : NEGF;
      mx = fmaxf(mx, v);
    }
    float s = 0.0f;
    for (int kk = 0; kk < 192; ++kk) {
      const int rel = kk - 64 - q;
      const int jab = blk * 64 + kk - 64;
      const bool valid = (rel >= -32) && (rel <= 31) && (jab >= 0) && (jab < NSEQ);
      const float v = valid ? Llog[q][kk] : NEGF;
      const float e = __expf(v - mx);
      s += e;
      Llog[q][kk] = e;
    }
    const float inv = 1.0f / s;
    u16* Pr = (u16*)&Llog[q][0];            // bf16 view overwrites front of row
    for (int kk = 0; kk < 192; ++kk) Pr[kk] = f2bf(Llog[q][kk] * inv);
  } else {
    const int q = tid - 64;
    float mx = -3.4e38f;
    for (int g = 0; g < 64; ++g) mx = fmaxf(mx, Glog[q][g]);
    float s = 0.0f;
    for (int g = 0; g < 64; ++g) {
      const float e = __expf(Glog[q][g] - mx);
      s += e;
      Glog[q][g] = e;
    }
    const float inv = 1.0f / s;
    u16* Pr = (u16*)&Glog[q][0];
    for (int g = 0; g < 64; ++g) Pr[g] = f2bf(Glog[q][g] * inv);
  }

  __syncthreads();

  // --- (P @ V) + (Pg @ Vg): 16 x 64 output strip per wave ---
  v8f o[4];
#pragma unroll
  for (int j = 0; j < 4; ++j) o[j] = zeroacc();

  const u16* Vbase = Vt + bh * DH * NSEQ;
#pragma unroll
  for (int ks = 0; ks < 6; ++ks) {          // 192 = 6 * 32 local keys
    const u16* Pr = (const u16*)&Llog[q0 + nidx][0] + ks * 32;
    const v16bf a = load_fragA(Pr, half);
    const int c0 = blk * 64 - 64 + ks * 32 + half * 16;   // 16 consecutive tokens
    const bool valid = (c0 >= 0) && (c0 < NSEQ);
#pragma unroll
    for (int j = 0; j < 4; ++j) {
      const int d = j * 16 + nidx;
      const u16* vp = Vbase + (size_t)d * NSEQ + blk * 64 - 64 + ks * 32;
      const v16bf bf = load_fragB_guard(vp, half, valid);
      o[j] = wmma_bf16(a, bf, o[j]);
    }
  }
  const u16* Vgbase = Vgt + bh * DH * NB;
#pragma unroll
  for (int ks = 0; ks < 2; ++ks) {          // 64 global keys
    const u16* Pr = (const u16*)&Glog[q0 + nidx][0] + ks * 32;
    const v16bf a = load_fragA(Pr, half);
#pragma unroll
    for (int j = 0; j < 4; ++j) {
      const int d = j * 16 + nidx;
      const v16bf bf = load_fragB(Vgbase + (size_t)d * NB + ks * 32, half);
      o[j] = wmma_bf16(a, bf, o[j]);
    }
  }

#pragma unroll
  for (int j = 0; j < 4; ++j)
#pragma unroll
    for (int r = 0; r < 8; ++r) {
      const int nn = blk * 64 + q0 + r + 8 * half;
      const int c = h * 64 + j * 16 + nidx;
      AO[((size_t)bb * NSEQ + nn) * CDIM + c] = f2bf(o[j][r]);
    }
}

// ---------------- stage 3: output GEMM + bias ----------------
// same LDS-staged 64x256-strip structure as the QKV GEMM

__global__ __launch_bounds__(128) void k_gemm_out(
    const u16* __restrict__ AO, const u16* __restrict__ Wot,
    const float* __restrict__ bias, float* __restrict__ out) {
  __shared__ u16 Atile[2][64 * 32];

  const int bx = blockIdx.x;               // 128 * 4
  const int mg = bx >> 2, ng = bx & 3;
  const int tid = threadIdx.x;
  const int wave = tid >> 5, lane = tid & 31;
  const int half = lane >> 4, nidx = lane & 15;
  const int m0 = mg * 64;
  const int n0 = ng * 256 + wave * 64;

  v8f acc[4][4];
#pragma unroll
  for (int i = 0; i < 4; ++i)
#pragma unroll
    for (int j = 0; j < 4; ++j) acc[i][j] = zeroacc();

  auto stage = [&](int buf, int kb) {
#pragma unroll
    for (int q = 0; q < 2; ++q) {
      const int chunk = tid + q * 128;
      const int row = chunk >> 2, cc = chunk & 3;
      stage_chunk16(AO + (size_t)(m0 + row) * CDIM + kb + cc * 8,
                    &Atile[buf][chunk * 8]);
    }
  };

  stage(0, 0);
  wait_stage();
  __syncthreads();

  for (int kb = 0; kb < CDIM; kb += 32) {
    const int buf = (kb >> 5) & 1;
    if (kb + 32 < CDIM) stage(buf ^ 1, kb + 32);

    v16bf a[4], b[4];
#pragma unroll
    for (int i = 0; i < 4; ++i)
      a[i] = load_fragA(&Atile[buf][(i * 16 + nidx) * 32], half);
#pragma unroll
    for (int j = 0; j < 4; ++j)
      b[j] = load_fragB(Wot + (size_t)(n0 + j * 16 + nidx) * CDIM + kb, half);
#pragma unroll
    for (int i = 0; i < 4; ++i)
#pragma unroll
      for (int j = 0; j < 4; ++j) acc[i][j] = wmma_bf16(a[i], b[j], acc[i][j]);

    wait_stage();
    __syncthreads();
  }

#pragma unroll
  for (int i = 0; i < 4; ++i)
#pragma unroll
    for (int j = 0; j < 4; ++j) {
      const int c = n0 + j * 16 + nidx;
      const float bv = bias[c];
#pragma unroll
      for (int r = 0; r < 8; ++r) {
        const int m = m0 + i * 16 + r + 8 * half;
        out[(size_t)m * CDIM + c] = acc[i][j][r] + bv;
      }
    }
}

// ---------------- launch ----------------

extern "C" void kernel_launch(void* const* d_in, const int* in_sizes, int n_in,
                              void* d_out, int out_size, void* d_ws, size_t ws_size,
                              hipStream_t stream) {
  const float* x    = (const float*)d_in[0];
  const float* wqkv = (const float*)d_in[1];
  const float* wout = (const float*)d_in[2];
  const float* bout = (const float*)d_in[3];
  float* out = (float*)d_out;

  char* ws = (char*)d_ws;
  size_t off = 0;
  auto carve = [&](size_t bytes) -> char* {
    char* p = ws + off;
    off += (bytes + 255) & ~(size_t)255;
    return p;
  };
  u16* Xb  = (u16*)carve((size_t)TOK * CDIM * 2);
  u16* Wqt = (u16*)carve((size_t)C3 * CDIM * 2);
  u16* Wot = (u16*)carve((size_t)CDIM * CDIM * 2);
  u16* Qb  = (u16*)carve((size_t)BATCH * HEADS * NSEQ * DH * 2);
  u16* Kb  = (u16*)carve((size_t)BATCH * HEADS * NSEQ * DH * 2);
  u16* Vt  = (u16*)carve((size_t)BATCH * HEADS * DH * NSEQ * 2);
  u16* Vgt = (u16*)carve((size_t)BATCH * HEADS * DH * NB * 2);
  u16* AO  = (u16*)carve((size_t)TOK * CDIM * 2);

  k_f2bf<<<2048, 256, 0, stream>>>(x, Xb, (long)TOK * CDIM);
  k_tr_f2bf<<<2048, 256, 0, stream>>>(wqkv, Wqt, C3, CDIM);
  k_tr_f2bf<<<2048, 256, 0, stream>>>(wout, Wot, CDIM, CDIM);
  k_gemm_qkv<<<128 * 12, 128, 0, stream>>>(Xb, Wqt, Qb, Kb, Vt, Vgt);
  k_attn<<<BATCH * HEADS * NB, 128, 0, stream>>>(Qb, Kb, Vt, Vgt, AO);
  k_gemm_out<<<128 * 4, 128, 0, stream>>>(AO, Wot, bout, out);
}
